// VQEmbedding_58918361366708
// MI455X (gfx1250) — matrix operand order repaired and checked by reading, not compiled, not run on hardware
//
#include <hip/hip_runtime.h>
#include <hip/hip_bf16.h>
#include <stdint.h>

typedef __bf16 v16bf __attribute__((ext_vector_type(16)));
typedef __bf16 v4bf  __attribute__((ext_vector_type(4)));
typedef float  v8f   __attribute__((ext_vector_type(8)));

#define DECAYF 0.99f
#define OMDF   0.01f
#define EPSF   1e-5f

constexpr int Dz = 64;
constexpr int Mz = 8192;
constexpr int Nz = 32 * 1024;      // B*T
constexpr int CHUNK = 256;         // codebook rows staged in LDS per pass
constexpr int ROWS_PER_WG = 128;   // 8 waves * 16 rows

// ---------------- zero scratch accumulators (counts + dw, contiguous) -------
__global__ void k_zero(float4* p) {
    int t = blockIdx.x * blockDim.x + threadIdx.x;   // 520*256 = 133120 float4
    float4 z; z.x = 0.f; z.y = 0.f; z.z = 0.f; z.w = 0.f;
    p[t] = z;
}

// ---------------- e2[m] = ||emb_m||^2 (exact fp32) --------------------------
__global__ void k_norm(const float* __restrict__ emb, float* __restrict__ e2) {
    int m = blockIdx.x * blockDim.x + threadIdx.x;
    const float4* row = (const float4*)(emb + (size_t)m * Dz);
    float s = 0.f;
#pragma unroll
    for (int k = 0; k < Dz / 4; ++k) {
        float4 v = row[k];
        s += v.x * v.x + v.y * v.y + v.z * v.z + v.w * v.w;
    }
    e2[m] = s;
}

// ---------------- fused distance + argmin via bf16x3 WMMA -------------------
__global__ __launch_bounds__(256)
void k_argmin(const float* __restrict__ flat, const float* __restrict__ emb,
              const float* __restrict__ e2, int* __restrict__ idx) {
    __shared__ __bf16 lds_hi[CHUNK * Dz];   // 32 KB
    __shared__ __bf16 lds_lo[CHUNK * Dz];   // 32 KB

    const int tid  = threadIdx.x;
    const int lane = tid & 31;
    const int wave = tid >> 5;
    const int rowbase = blockIdx.x * ROWS_PER_WG + wave * 16;
    const int rowg = rowbase + (lane & 15);
    const int kbA  = (lane < 16) ? 0 : 16;   // 16-bit A-fragment K base per half-wave
    const int kbB  = (lane < 16) ? 0 : 16;   // 16-bit B-fragment K base per half-wave

    // A fragment: 16 rows x 64 K of flat, split into bf16 hi + lo residual.
    v16bf a0h, a0l, a1h, a1l;
    {
        const float* fr = flat + (size_t)rowg * Dz;
#pragma unroll
        for (int e = 0; e < 16; ++e) {
            int k = kbA + e;
            float f0 = fr[k];
            __bf16 h0 = (__bf16)f0;
            a0h[e] = h0; a0l[e] = (__bf16)(f0 - (float)h0);
            float f1 = fr[k + 32];
            __bf16 h1 = (__bf16)f1;
            a1h[e] = h1; a1l[e] = (__bf16)(f1 - (float)h1);
        }
    }

    float    bestd[8];
    uint32_t besti[8];
#pragma unroll
    for (int r = 0; r < 8; ++r) { bestd[r] = 3.0e38f; besti[r] = 0u; }

    for (int chunk = 0; chunk < Mz / CHUNK; ++chunk) {
        const int mbase = chunk * CHUNK;
        // cooperatively stage codebook chunk into LDS as bf16 hi/lo planes
        const float4* src = (const float4*)(emb + (size_t)mbase * Dz);
        for (int it = tid; it < CHUNK * (Dz / 4); it += 256) {
            float4 v = src[it];
            v4bf h, l;
            h[0] = (__bf16)v.x; l[0] = (__bf16)(v.x - (float)h[0]);
            h[1] = (__bf16)v.y; l[1] = (__bf16)(v.y - (float)h[1]);
            h[2] = (__bf16)v.z; l[2] = (__bf16)(v.z - (float)h[2]);
            h[3] = (__bf16)v.w; l[3] = (__bf16)(v.w - (float)h[3]);
            *(v4bf*)&lds_hi[it * 4] = h;
            *(v4bf*)&lds_lo[it * 4] = l;
        }
        __syncthreads();

        for (int t = 0; t < CHUNK / 16; ++t) {
            const int      ml  = t * 16 + (lane & 15);
            const uint32_t mg  = (uint32_t)(mbase + t * 16 + (lane & 15));
            const float    e2v = e2[mg];
            const v16bf b0h = *(const v16bf*)&lds_hi[ml * Dz + kbB];
            const v16bf b1h = *(const v16bf*)&lds_hi[ml * Dz + 32 + kbB];
            const v16bf b0l = *(const v16bf*)&lds_lo[ml * Dz + kbB];
            const v16bf b1l = *(const v16bf*)&lds_lo[ml * Dz + 32 + kbB];

            v8f c = {};
            // c = A*B with A = (hi+lo), B = (hi+lo), dropping lo*lo (~2^-16 rel)
            c = __builtin_amdgcn_wmma_f32_16x16x32_bf16(false, a0h, false, b0h, (short)0, c, false, false);
            c = __builtin_amdgcn_wmma_f32_16x16x32_bf16(false, a1h, false, b1h, (short)0, c, false, false);
            c = __builtin_amdgcn_wmma_f32_16x16x32_bf16(false, a0l, false, b0h, (short)0, c, false, false);
            c = __builtin_amdgcn_wmma_f32_16x16x32_bf16(false, a1l, false, b1h, (short)0, c, false, false);
            c = __builtin_amdgcn_wmma_f32_16x16x32_bf16(false, a0h, false, b0l, (short)0, c, false, false);
            c = __builtin_amdgcn_wmma_f32_16x16x32_bf16(false, a1h, false, b1l, (short)0, c, false, false);

#pragma unroll
            for (int r = 0; r < 8; ++r) {
                float d  = fmaf(-2.0f, c[r], e2v);   // ||f||^2 constant per row: drop it
                bool  lt = d < bestd[r];
                bestd[r] = lt ? d  : bestd[r];
                besti[r] = lt ? mg : besti[r];
            }
        }
        __syncthreads();
    }

    // min-reduce across the 16 lanes of each half-wave (xor butterfly)
#pragma unroll
    for (int r = 0; r < 8; ++r) {
        float    d = bestd[r];
        uint32_t i = besti[r];
#pragma unroll
        for (int mask = 1; mask <= 8; mask <<= 1) {
            float    od = __shfl_xor(d, mask, 32);
            uint32_t oi = __shfl_xor(i, mask, 32);
            if (od < d || (od == d && oi < i)) { d = od; i = oi; }
        }
        if ((lane & 15) == 0)
            idx[rowbase + r + 8 * (lane >> 4)] = (int)i;
    }
}

// ---------------- gather quantized + histogram + segment-sum ----------------
__global__ void k_scatter(const float* __restrict__ flat, const float* __restrict__ emb,
                          const int* __restrict__ idx, float* __restrict__ q_out,
                          float* __restrict__ counts, float* __restrict__ dw) {
    int t = blockIdx.x * blockDim.x + threadIdx.x;   // N*16 threads, float4 each
    int n = t >> 4, j = t & 15;
    int m = idx[n];
    float4 f = ((const float4*)flat)[(size_t)n * 16 + j];
    float4 q = ((const float4*)emb)[(size_t)m * 16 + j];
    ((float4*)q_out)[(size_t)n * 16 + j] = q;        // straight-through value = emb[idx]
    float* dp = dw + (size_t)m * Dz + j * 4;
    atomicAdd(dp + 0, f.x);
    atomicAdd(dp + 1, f.y);
    atomicAdd(dp + 2, f.z);
    atomicAdd(dp + 3, f.w);
    if (j == 0) atomicAdd(counts + m, 1.0f);
}

// ---------------- EMA count update + Laplace renorm (single block) ----------
__global__ __launch_bounds__(1024)
void k_count(const float* __restrict__ old_cnt, const float* __restrict__ counts,
             float* __restrict__ emac_out, float* __restrict__ emac_ws) {
    __shared__ float red[1024];
    int t = threadIdx.x;
    float raw[8]; float s = 0.f;
#pragma unroll
    for (int k = 0; k < 8; ++k) {
        int m = t + k * 1024;
        float r = DECAYF * old_cnt[m] + OMDF * counts[m];
        raw[k] = r; s += r;
    }
    red[t] = s;
    __syncthreads();
    for (int off = 512; off > 0; off >>= 1) {
        if (t < off) red[t] += red[t + off];
        __syncthreads();
    }
    float nsum  = red[0];
    float scale = nsum / (nsum + (float)Mz * EPSF);
#pragma unroll
    for (int k = 0; k < 8; ++k) {
        int m = t + k * 1024;
        float v = (raw[k] + EPSF) * scale;
        emac_out[m] = v;
        emac_ws[m]  = v;
    }
}

// ---------------- EMA embedding + normalized codebook -----------------------
__global__ void k_embfin(const float* __restrict__ dw, const float* __restrict__ old_ema,
                         const float* __restrict__ emac, float* __restrict__ newemb_o,
                         float* __restrict__ emae_o) {
    int t = blockIdx.x * blockDim.x + threadIdx.x;   // M*16 threads, float4 each
    int m = t >> 4;
    float4 d = ((const float4*)dw)[t];
    float4 o = ((const float4*)old_ema)[t];
    float4 e;
    e.x = o.x * DECAYF + OMDF * d.x;
    e.y = o.y * DECAYF + OMDF * d.y;
    e.z = o.z * DECAYF + OMDF * d.z;
    e.w = o.w * DECAYF + OMDF * d.w;
    ((float4*)emae_o)[t] = e;
    float inv = 1.0f / emac[m];
    float4 nv; nv.x = e.x * inv; nv.y = e.y * inv; nv.z = e.z * inv; nv.w = e.w * inv;
    ((float4*)newemb_o)[t] = nv;
}

extern "C" void kernel_launch(void* const* d_in, const int* in_sizes, int n_in,
                              void* d_out, int out_size, void* d_ws, size_t ws_size,
                              hipStream_t stream) {
    (void)in_sizes; (void)n_in; (void)out_size; (void)ws_size;
    const float* flat    = (const float*)d_in[0];  // eeg_semantic  [N,D]
    const float* emb     = (const float*)d_in[1];  // eeg_embedding [M,D]
    const float* old_cnt = (const float*)d_in[2];  // eeg_ema_count [M]
    const float* old_ema = (const float*)d_in[3];  // eeg_ema_embedding [M,D]

    char*  ws      = (char*)d_ws;
    int*   idx     = (int*)  (ws + 0);                       // N ints      (128 KB)
    float* counts  = (float*)(ws + 131072);                  // M floats    ( 32 KB)
    float* dw      = (float*)(ws + 163840);                  // M*D floats  (  2 MB) (contiguous after counts)
    float* e2      = (float*)(ws + 2260992);                 // M floats
    float* emac_ws = (float*)(ws + 2293760);                 // M floats

    float* out      = (float*)d_out;
    float* q_out    = out;                                   // [N,D]
    float* newemb_o = out + (size_t)Nz * Dz;                 // [M,D]
    float* emac_o   = newemb_o + (size_t)Mz * Dz;            // [M]
    float* emae_o   = emac_o + Mz;                           // [M,D]

    k_zero   <<<520, 256, 0, stream>>>((float4*)counts);     // counts+dw = 133120 float4
    k_norm   <<<Mz / 256, 256, 0, stream>>>(emb, e2);
    k_argmin <<<Nz / ROWS_PER_WG, 256, 0, stream>>>(flat, emb, e2, idx);
    k_scatter<<<(Nz * 16) / 256, 256, 0, stream>>>(flat, emb, idx, q_out, counts, dw);
    k_count  <<<1, 1024, 0, stream>>>(old_cnt, counts, emac_o, emac_ws);
    k_embfin <<<(Mz * 16) / 256, 256, 0, stream>>>(dw, old_ema, emac_ws, newemb_o, emae_o);
}
